// GraphUNet_14242111553929
// MI455X (gfx1250) — 8-bit matrix operand order repaired and checked by reading, compiled, not run on hardware
//
#include <hip/hip_runtime.h>
#include <math.h>

typedef float v2f __attribute__((ext_vector_type(2)));
typedef float v8f __attribute__((ext_vector_type(8)));
typedef int   v8i __attribute__((ext_vector_type(8)));

#define N0   4096
#define CDIM 256
#define BM 32
#define BN 64
#define BK 32
#define LDAS 36
#define LDBS 72

#define ASYNC_B128(ldsaddr, gaddr)                                   \
  asm volatile("global_load_async_to_lds_b128 %0, %1, off"           \
               :: "v"(ldsaddr), "v"(gaddr) : "memory")
#define WAIT_ASYNC() asm volatile("s_wait_asynccnt 0x0" ::: "memory")

static __device__ __forceinline__ unsigned ldsAddr(const void* p) {
  return (unsigned)(size_t)p;
}


__global__ void k_zero(float* __restrict__ p, long n) {
  long i = (long)blockIdx.x * blockDim.x + threadIdx.x;
  if (i < n) p[i] = 0.0f;
}

__global__ void k_scatter_edges(const int* __restrict__ ei, int E,
                                float* __restrict__ A, int n) {
  int e = blockIdx.x * blockDim.x + threadIdx.x;
  if (e < E) {
    int s = ei[e];
    int d = ei[E + e];
    atomicAdd(&A[(size_t)s * n + d], 1.0f);
  }
}

__global__ void k_f2u8(const float* __restrict__ A, unsigned char* __restrict__ A8, long nn) {
  long i = (long)blockIdx.x * 256 + threadIdx.x;
  if (i < nn) A8[i] = (unsigned char)fminf(A[i], 255.0f);
}

__global__ void k_deg_dis(const float* __restrict__ A, int n, float* __restrict__ dis) {
  __shared__ float red[256];
  int row = blockIdx.x;
  const float* Ar = A + (size_t)row * n;
  float s = 0.0f;
  for (int c = threadIdx.x; c < n; c += 256) s += Ar[c];
  red[threadIdx.x] = s; __syncthreads();
  for (int st = 128; st > 0; st >>= 1) {
    if (threadIdx.x < st) red[threadIdx.x] += red[threadIdx.x + st];
    __syncthreads();
  }
  if (threadIdx.x == 0) {
    float deg = red[0] + 2.0f;
    dis[row] = (deg > 0.0f) ? rsqrtf(fmaxf(deg, 1e-12f)) : 0.0f;
  }
}

__global__ void k_scale_rows(const float* __restrict__ y, const float* __restrict__ dis,
                             float* __restrict__ yp) {
  int r = blockIdx.x;
  yp[(size_t)r * CDIM + threadIdx.x] = dis[r] * y[(size_t)r * CDIM + threadIdx.x];
}

__global__ void k_pnorm(const float* __restrict__ p, float* __restrict__ pinv) {
  __shared__ float red[256];
  float v = p[threadIdx.x];
  red[threadIdx.x] = v * v; __syncthreads();
  for (int st = 128; st > 0; st >>= 1) {
    if (threadIdx.x < st) red[threadIdx.x] += red[threadIdx.x + st];
    __syncthreads();
  }
  if (threadIdx.x == 0) pinv[0] = rsqrtf(fmaxf(red[0], 1e-30f));
}

__global__ void k_score(const float* __restrict__ x, const float* __restrict__ p,
                        const float* __restrict__ pinv, float* __restrict__ score) {
  __shared__ float red[256];
  int row = blockIdx.x;
  red[threadIdx.x] = x[(size_t)row * CDIM + threadIdx.x] * p[threadIdx.x];
  __syncthreads();
  for (int st = 128; st > 0; st >>= 1) {
    if (threadIdx.x < st) red[threadIdx.x] += red[threadIdx.x + st];
    __syncthreads();
  }
  if (threadIdx.x == 0) score[row] = red[0] * pinv[0];
}

__global__ __launch_bounds__(1024) void k_topk(const float* __restrict__ score, int n, int k,
                                               float* __restrict__ sv, int* __restrict__ perm) {
  __shared__ float ss[4096];
  __shared__ int   si[4096];
  int tid = threadIdx.x;
  for (int i = tid; i < n; i += 1024) { ss[i] = score[i]; si[i] = i; }
  __syncthreads();
  for (int sz = 2; sz <= n; sz <<= 1) {
    for (int j = sz >> 1; j > 0; j >>= 1) {
      for (int i = tid; i < n; i += 1024) {
        int ixj = i ^ j;
        if (ixj > i) {
          float sa = ss[i], sb = ss[ixj];
          int ia = si[i], ib = si[ixj];
          bool aFirst = (sa > sb) || (sa == sb && ia < ib);
          bool desc = ((i & sz) == 0);
          if (desc ? !aFirst : aFirst) {
            ss[i] = sb; ss[ixj] = sa; si[i] = ib; si[ixj] = ia;
          }
        }
      }
      __syncthreads();
    }
  }
  for (int i = tid; i < k; i += 1024) { sv[i] = ss[i]; perm[i] = si[i]; }
}

__global__ void k_gather_gate(const float* __restrict__ xin, const int* __restrict__ perm,
                              const float* __restrict__ sv, float* __restrict__ xout) {
  int i = blockIdx.x;
  float g = tanhf(sv[i]);
  xout[(size_t)i * CDIM + threadIdx.x] = xin[(size_t)perm[i] * CDIM + threadIdx.x] * g;
}

__global__ void k_scatter_add(float* __restrict__ dst, const float* __restrict__ src,
                              const int* __restrict__ perm) {
  int i = blockIdx.x;
  dst[(size_t)perm[i] * CDIM + threadIdx.x] += src[(size_t)i * CDIM + threadIdx.x];
}

__global__ void k_make_R(const unsigned char* __restrict__ A8, const int* __restrict__ perm,
                         int n, unsigned char* __restrict__ R) {
  int i = blockIdx.x;
  int pi = perm[i];
  int c16 = threadIdx.x << 4;
  if (c16 < n) {
    uint4 v = *(const uint4*)(A8 + (size_t)pi * n + c16);
    if (pi >= c16 && pi < c16 + 16) ((unsigned char*)&v)[pi - c16] += 1;
    *(uint4*)(R + (size_t)i * n + c16) = v;
  }
}

__global__ void k_make_Gt(const unsigned char* __restrict__ A8, const int* __restrict__ perm,
                          int n, unsigned char* __restrict__ Gt) {
  int j = blockIdx.x;
  int pj = perm[j];
  for (int kk = threadIdx.x; kk < n; kk += 256) {
    unsigned char v = A8[(size_t)kk * n + pj];
    if (kk == pj) v = (unsigned char)(v + 1);
    Gt[(size_t)j * n + kk] = v;
  }
}

__global__ __launch_bounds__(256)
void k_aug_u8(const unsigned char* __restrict__ R, const unsigned char* __restrict__ Gt,
              float* __restrict__ Aout, unsigned char* __restrict__ Aout8,
              int kdim, int n) {
  __shared__ unsigned int A8s[32 * 36];
  __shared__ unsigned int Bts[64 * 36];

  const int tid = threadIdx.x;
  const int lane = tid & 31, wave = tid >> 5;
  const int wy = wave >> 2, wx = wave & 3;
  const int m16 = lane & 15, h = lane >> 4;
  const int bm = blockIdx.y * 32, bn = blockIdx.x * 64;

  v8i acc = {};
  for (int k0 = 0; k0 < n; k0 += 128) {
    __syncthreads();
    {
      int r = tid >> 3, c16 = (tid & 7) << 4;
      ASYNC_B128(ldsAddr(&A8s[r * 36 + (c16 >> 2)]),
                 R + (size_t)(bm + r) * n + k0 + c16);
    }
    {
      int j = tid >> 2, c32 = (tid & 3) << 5;
      const unsigned char* src = Gt + (size_t)(bn + j) * n + k0 + c32;
      ASYNC_B128(ldsAddr(&Bts[j * 36 + (c32 >> 2)]), src);
      ASYNC_B128(ldsAddr(&Bts[j * 36 + (c32 >> 2) + 4]), src + 16);
    }
    WAIT_ASYNC();
    __syncthreads();

    const unsigned int* Ar = &A8s[(wy * 16 + m16) * 36];
    const unsigned int* Br = &Bts[(wx * 16 + m16) * 36];
    #pragma unroll
    for (int s = 0; s < 2; ++s) {
      v8i a, b;
      int ab = s * 16 + 2 * h;
      a[0] = Ar[ab + 0];  a[1] = Ar[ab + 1];
      a[2] = Ar[ab + 4];  a[3] = Ar[ab + 5];
      a[4] = Ar[ab + 8];  a[5] = Ar[ab + 9];
      a[6] = Ar[ab + 12]; a[7] = Ar[ab + 13];
      int bb = s * 16 + 2 * h;
      b[0] = Br[bb + 0];  b[1] = Br[bb + 1];  b[2] = Br[bb + 4];  b[3] = Br[bb + 5];
      b[4] = Br[bb + 8];  b[5] = Br[bb + 9];  b[6] = Br[bb + 12]; b[7] = Br[bb + 13];
      acc = __builtin_amdgcn_wmma_i32_16x16x64_iu8(false, a, false, b, acc, false, false);
    }
  }
  #pragma unroll
  for (int r = 0; r < 8; ++r) {
    int row = bm + wy * 16 + r + 8 * h;
    int col = bn + wx * 16 + m16;
    int v = (row == col) ? 0 : acc[r];
    Aout[(size_t)row * kdim + col] = (float)v;
    Aout8[(size_t)row * kdim + col] = (unsigned char)(v > 255 ? 255 : v);
  }
}

template<int MODE, bool RELU>
__global__ __launch_bounds__(256)
void k_wmma_gemm(const float* __restrict__ Ag, const float* __restrict__ Bg,
                 float* __restrict__ Cg,
                 int M, int N, int K, int lda, int ldb, int ldc,
                 const float* __restrict__ dis, const float* __restrict__ bias) {
  __shared__ float Asl[BM][LDAS];
  __shared__ float Bsl[BK][LDBS];

  const int tid  = threadIdx.x;
  const int lane = tid & 31;
  const int wave = tid >> 5;
  const int wy   = wave >> 2;
  const int wx   = wave & 3;
  const int m16  = lane & 15;
  const int h    = lane >> 4;
  const int bm = blockIdx.y * BM;
  const int bn = blockIdx.x * BN;

  v8f acc = {};

  for (int k0 = 0; k0 < K; k0 += BK) {
    __syncthreads();
    {
      int r = tid >> 3, c4 = (tid & 7) << 2;
      ASYNC_B128(ldsAddr(&Asl[r][c4]), Ag + (size_t)(bm + r) * lda + k0 + c4);
    }
    {
      int r0 = tid >> 4, c4 = (tid & 15) << 2;
      ASYNC_B128(ldsAddr(&Bsl[r0][c4]),      Bg + (size_t)(k0 + r0) * ldb + bn + c4);
      ASYNC_B128(ldsAddr(&Bsl[r0 + 16][c4]), Bg + (size_t)(k0 + r0 + 16) * ldb + bn + c4);
    }
    WAIT_ASYNC();
    __syncthreads();

    if (k0 + BK < K) {
      __builtin_prefetch(Ag + (size_t)(bm + (tid >> 3)) * lda + k0 + BK, 0, 1);
    }
    #pragma unroll
    for (int kk = 0; kk < BK; kk += 4) {
      v2f a = *(const v2f*)&Asl[wy * 16 + m16][kk + 2 * h];
      v2f b;
      b.x = Bsl[kk + 2 * h][wx * 16 + m16];
      b.y = Bsl[kk + 2 * h + 1][wx * 16 + m16];
      acc = __builtin_amdgcn_wmma_f32_16x16x4_f32(
          false, a, false, b, (short)0, acc, false, false);
    }
  }
  #pragma unroll
  for (int r = 0; r < 8; ++r) {
    int row = bm + wy * 16 + r + 8 * h;
    int col = bn + wx * 16 + m16;
    float v = acc[r];
    if (MODE == 1) {
      v = dis[row] * (v + 2.0f * Bg[(size_t)row * ldb + col]) + bias[col];
      if (RELU) v = fmaxf(v, 0.0f);
    }
    Cg[(size_t)row * ldc + col] = v;
  }
}


static inline void gemm_plain(const float* A, const float* B, float* Cg,
                              int M, int K, hipStream_t s) {
  dim3 g(CDIM / BN, M / BM), b(256);
  k_wmma_gemm<0, false><<<g, b, 0, s>>>(A, B, Cg, M, CDIM, K, K, CDIM, CDIM,
                                        nullptr, nullptr);
}

static inline void gemm_agg(const float* Adj, const float* yp, float* Cg, int n,
                            const float* dis, const float* bias, bool relu,
                            hipStream_t s) {
  dim3 g(CDIM / BN, n / BM), b(256);
  if (relu)
    k_wmma_gemm<1, true ><<<g, b, 0, s>>>(Adj, yp, Cg, n, CDIM, n, n, CDIM, CDIM, dis, bias);
  else
    k_wmma_gemm<1, false><<<g, b, 0, s>>>(Adj, yp, Cg, n, CDIM, n, n, CDIM, CDIM, dis, bias);
}

extern "C" void kernel_launch(void* const* d_in, const int* in_sizes, int n_in,
                              void* d_out, int out_size, void* d_ws, size_t ws_size,
                              hipStream_t stream) {
  (void)n_in; (void)out_size; (void)ws_size;
  const float* x_in    = (const float*)d_in[0];
  const int*   ei      = (const int*)  d_in[1];
  const float* W_init  = (const float*)d_in[2];
  const float* b_init  = (const float*)d_in[3];
  const float* W_down  = (const float*)d_in[4];
  const float* b_down  = (const float*)d_in[5];
  const float* p_pool  = (const float*)d_in[6];
  const float* W_up    = (const float*)d_in[7];
  const float* b_up    = (const float*)d_in[8];
  const float* W_final = (const float*)d_in[9];
  const float* b_final = (const float*)d_in[10];
  float* out = (float*)d_out;
  const int E = in_sizes[1] / 2;

  size_t off = 0;
  auto alloc = [&](size_t bytes) -> void* {
    void* p = (char*)d_ws + off;
    off += (bytes + 255) & ~(size_t)255;
    return p;
  };
  float* A0    = (float*)alloc((size_t)4096 * 4096 * 4);
  float* A1    = (float*)alloc((size_t)2048 * 2048 * 4);
  float* A2    = (float*)alloc((size_t)1024 * 1024 * 4);
  float* A3    = (float*)alloc((size_t)512  * 512  * 4);
  unsigned char* A0u8 = (unsigned char*)alloc((size_t)4096 * 4096);
  unsigned char* A1u8 = (unsigned char*)alloc((size_t)2048 * 2048);
  unsigned char* A2u8 = (unsigned char*)alloc((size_t)1024 * 1024);
  unsigned char* A3u8 = (unsigned char*)alloc((size_t)512 * 512);
  unsigned char* Rbuf = (unsigned char*)alloc((size_t)2048 * 4096);
  unsigned char* Gtb  = (unsigned char*)alloc((size_t)2048 * 4096);
  float* xs0  = (float*)alloc((size_t)4096 * CDIM * 4);
  float* xs1  = (float*)alloc((size_t)2048 * CDIM * 4);
  float* xs2  = (float*)alloc((size_t)1024 * CDIM * 4);
  float* xg   = (float*)alloc((size_t)2048 * CDIM * 4);
  float* xb   = (float*)alloc((size_t)2048 * CDIM * 4);
  float* ybuf = (float*)alloc((size_t)4096 * CDIM * 4);
  float* yp   = (float*)alloc((size_t)4096 * CDIM * 4);
  float* dis  = (float*)alloc(4096 * 4);
  float* scr  = (float*)alloc(4096 * 4);
  float* sv   = (float*)alloc(4096 * 4);
  float* pinv = (float*)alloc(256);
  int* perm0  = (int*)alloc(2048 * 4);
  int* perm1  = (int*)alloc(1024 * 4);
  int* perm2  = (int*)alloc(512 * 4);

  k_zero<<<(4096L * 4096 + 255) / 256, 256, 0, stream>>>(A0, 4096L * 4096);
  k_scatter_edges<<<(E + 255) / 256, 256, 0, stream>>>(ei, E, A0, N0);
  k_f2u8<<<(int)((4096L * 4096 + 255) / 256), 256, 0, stream>>>(A0, A0u8, 4096L * 4096);

  gemm_plain(x_in, W_init, ybuf, N0, CDIM, stream);
  k_deg_dis<<<N0, 256, 0, stream>>>(A0, N0, dis);
  k_scale_rows<<<N0, 256, 0, stream>>>(ybuf, dis, yp);
  gemm_agg(A0, yp, xs0, N0, dis, b_init, false, stream);

  float*         Aall[4]  = {A0, A1, A2, A3};
  unsigned char* A8all[4] = {A0u8, A1u8, A2u8, A3u8};
  float*         xsAll[3] = {xs0, xs1, xs2};
  float*         dOut[3]  = {xs1, xs2, xb};
  int*           perms[3] = {perm0, perm1, perm2};
  for (int i = 0; i < 3; ++i) {
    int n = N0 >> i, k = n >> 1;
    const float* pvec = p_pool + i * CDIM;
    k_pnorm<<<1, 256, 0, stream>>>(pvec, pinv);
    k_score<<<n, 256, 0, stream>>>(xsAll[i], pvec, pinv, scr);
    k_topk<<<1, 1024, 0, stream>>>(scr, n, k, sv, perms[i]);
    k_gather_gate<<<k, 256, 0, stream>>>(xsAll[i], perms[i], sv, xg);
    k_make_R <<<k, 256, 0, stream>>>(A8all[i], perms[i], n, Rbuf);
    k_make_Gt<<<k, 256, 0, stream>>>(A8all[i], perms[i], n, Gtb);
    { dim3 g(k / 64, k / 32);
      k_aug_u8<<<g, 256, 0, stream>>>(Rbuf, Gtb, Aall[i + 1], A8all[i + 1], k, n); }
    gemm_plain(xg, W_down + (size_t)i * CDIM * CDIM, ybuf, k, CDIM, stream);
    k_deg_dis<<<k, 256, 0, stream>>>(Aall[i + 1], k, dis);
    k_scale_rows<<<k, 256, 0, stream>>>(ybuf, dis, yp);
    gemm_agg(Aall[i + 1], yp, dOut[i], k, dis, b_down + i * CDIM, true, stream);
  }

  k_scatter_add<<<512, 256, 0, stream>>>(xs2, xb, perm2);
  gemm_plain(xs2, W_up, ybuf, 1024, CDIM, stream);
  k_deg_dis<<<1024, 256, 0, stream>>>(A2, 1024, dis);
  k_scale_rows<<<1024, 256, 0, stream>>>(ybuf, dis, yp);
  gemm_agg(A2, yp, xb, 1024, dis, b_up, true, stream);

  k_scatter_add<<<1024, 256, 0, stream>>>(xs1, xb, perm1);
  gemm_plain(xs1, W_up + CDIM * CDIM, ybuf, 2048, CDIM, stream);
  k_deg_dis<<<2048, 256, 0, stream>>>(A1, 2048, dis);
  k_scale_rows<<<2048, 256, 0, stream>>>(ybuf, dis, yp);
  gemm_agg(A1, yp, xb, 2048, dis, b_up + CDIM, true, stream);

  k_scatter_add<<<2048, 256, 0, stream>>>(xs0, xb, perm0);

  gemm_plain(xs0, W_final, ybuf, N0, CDIM, stream);
  k_deg_dis<<<N0, 256, 0, stream>>>(A0, N0, dis);
  k_scale_rows<<<N0, 256, 0, stream>>>(ybuf, dis, yp);
  gemm_agg(A0, yp, out, N0, dis, b_final, false, stream);
}
